// Comapre_Aggregate_46969762349115
// MI455X (gfx1250) — compile-verified, matrix operand order checked by reading, not run on hardware
//
#include <hip/hip_runtime.h>
#include <hip/hip_bf16.h>
#include <math.h>
#include <stdint.h>

// ---------------------------------------------------------------------------
// Problem constants
// ---------------------------------------------------------------------------
constexpr int Bb = 8;          // batch
constexpr int L  = 2048;       // LQ == LA
constexpr int D  = 300;        // input feature dim
constexpr int Hh = 150;        // hidden dim
constexpr int DP = 320;        // D padded to multiple of 32 (K for gate GEMM)
constexpr int HP = 160;        // H padded to multiple of 16/32
constexpr int M  = Bb * L;     // 16384 total rows
constexpr int KT_D = DP / 32;  // 10 k-tiles for K=320
constexpr int KT_H = HP / 32;  // 5  k-tiles for K=160
constexpr int NT_H = HP / 16;  // 10 n-tiles for N=160

typedef __bf16 bf16;
typedef __attribute__((ext_vector_type(16))) __bf16 v16bf;
typedef __attribute__((ext_vector_type(8)))  float  v8f;
typedef __attribute__((ext_vector_type(4)))  unsigned int u32x4;

union FragU { u32x4 u4[2]; v16bf v; };

// ---------------------------------------------------------------------------
// Tensor Data Mover support (gfx1250). Builtin arity differs per toolchain:
// ROCm 7.2 / clang-22 -> 5 args ; amdgpu-toolchain / clang-23 -> 6 args.
// If the builtin is missing entirely we fall back to a cooperative copy.
// ---------------------------------------------------------------------------
#if __has_builtin(__builtin_amdgcn_tensor_load_to_lds)
#define HAVE_TDM 1
#else
#define HAVE_TDM 0
#endif

typedef __attribute__((ext_vector_type(4))) unsigned int tdmv4u;
typedef __attribute__((ext_vector_type(8))) int          tdmv8i;
typedef __attribute__((ext_vector_type(4))) int          tdmv4i;

__device__ inline void tdm_wait_le0() {
#if __has_builtin(__builtin_amdgcn_s_wait_tensorcnt)
  __builtin_amdgcn_s_wait_tensorcnt(0);
#else
  asm volatile("s_wait_tensorcnt 0x0" ::: "memory");
#endif
}
__device__ inline void tdm_wait_le1() {
#if __has_builtin(__builtin_amdgcn_s_wait_tensorcnt)
  __builtin_amdgcn_s_wait_tensorcnt(1);
#else
  asm volatile("s_wait_tensorcnt 0x1" ::: "memory");
#endif
}

#if HAVE_TDM
// DMA a 32-row x 160-col bf16 tile (rows contiguous, row stride HP elements)
// from global memory into LDS. 2D descriptor, groups 2/3 unused (zero).
__device__ inline void tdm_load_p_tile(const bf16* gsrc, void* lds_dst) {
  unsigned long long ga = (unsigned long long)(uintptr_t)gsrc;
  // Generic pointer to a __shared__ object: low 32 bits == LDS byte offset.
  unsigned lds_off = (unsigned)(uintptr_t)lds_dst;
  tdmv4u g0;
  g0[0] = 1u;                                            // count=1, user D#
  g0[1] = lds_off;                                       // lds_addr (bytes)
  g0[2] = (unsigned)ga;                                  // global_addr[31:0]
  g0[3] = (unsigned)((ga >> 32) & 0x01FFFFFFull)         // global_addr[56:32]
          | (2u << 30);                                  // type = 2 ("image")
  tdmv8i g1;
  g1[0] = 0x00010000;                                    // data_size = 2 bytes
  g1[1] = (int)((unsigned)(HP & 0xFFFF) << 16);          // tensor_dim0 = 160
  g1[2] = (int)((unsigned)(L  & 0xFFFF) << 16);          // tensor_dim1 = 2048
  g1[3] = (int)((unsigned)(HP & 0xFFFF) << 16);          // tile_dim0 = 160
  g1[4] = 32;                                            // tile_dim1 = 32 rows
  g1[5] = HP;                                            // dim0_stride = 160
  g1[6] = 0;
  g1[7] = 0;
  tdmv4i z = (tdmv4i)0;
#if __clang_major__ >= 23
  __builtin_amdgcn_tensor_load_to_lds(g0, g1, z, z, (tdmv8i)0, 0);
#else
  __builtin_amdgcn_tensor_load_to_lds(g0, g1, z, z, 0);
#endif
}
#endif

// ---------------------------------------------------------------------------
// Fragment loaders.
// A-fragment (16 rows x 32 k, bf16) per CDNA5 layout:
//   lane = hf*16 + r ; VGPR v holds K = (v&3)*2 + (v>>2)*16 + hf*8 (+0,+1)
// Rows are contiguous along k in memory -> two b128 loads per lane
// (global_load_b128 or ds_load_b128 depending on address space).
// ---------------------------------------------------------------------------
__device__ inline v16bf load_frag_rowmajor(const bf16* __restrict__ base, int ld) {
  int lane = threadIdx.x & 31;
  int r  = lane & 15;
  int hf = lane >> 4;
  const bf16* p = base + (size_t)r * ld + hf * 8;
  FragU f;
  f.u4[0] = *(const u32x4*)(p);       // k: hf*8 + 0..7
  f.u4[1] = *(const u32x4*)(p + 16);  // k: hf*8 + 16..23
  return f.v;
}

// Pre-packed weight fragment: 512 bf16 per (ntile,ktile), laid out so each
// lane reads its 16 elements as 32 contiguous bytes (2 x global_load_b128).
__device__ inline v16bf load_frag_packed(const bf16* __restrict__ tile) {
  int lane = threadIdx.x & 31;
  const u32x4* p = (const u32x4*)(tile + lane * 16);
  FragU f;
  f.u4[0] = p[0];
  f.u4[1] = p[1];
  return f.v;
}

__device__ inline v8f wmma_bf16(v16bf a, v16bf b, v8f c) {
  return __builtin_amdgcn_wmma_f32_16x16x32_bf16(
      /*neg_a=*/false, a, /*neg_b=*/false, b,
      /*c_mod=*/(short)0, c, /*reuse_a=*/false, /*reuse_b=*/false);
}

__device__ inline int klocal_of(int e, int hf) {
  int v = e >> 1, pos = e & 1;
  return ((v & 3) * 2) + ((v >> 2) * 16) + hf * 8 + pos;
}

// ---------------------------------------------------------------------------
// Kernel 1: fp32 -> bf16 with zero K-padding (D=300 -> 320)
// ---------------------------------------------------------------------------
__global__ void cmp_convert_pad_kernel(const float* __restrict__ in,
                                       bf16* __restrict__ out, int rows) {
  int idx = blockIdx.x * blockDim.x + threadIdx.x;
  int total = rows * DP;
  if (idx >= total) return;
  int r = idx / DP, k = idx - r * DP;
  float v = (k < D) ? in[(size_t)r * D + k] : 0.0f;
  out[idx] = (bf16)v;
}

// ---------------------------------------------------------------------------
// Kernel 2: pack Wi/Wu/Wg/Wt into WMMA B-fragment order (bf16, zero-padded).
// Wt is remapped for the padded concat layout [Ah 0..149 | pad | Hm 0..149 | pad].
// ---------------------------------------------------------------------------
__global__ void cmp_pack_weights_kernel(const float* __restrict__ Wi,
                                        const float* __restrict__ Wu,
                                        const float* __restrict__ Wg,
                                        const float* __restrict__ Wt,
                                        bf16* __restrict__ WiP, bf16* __restrict__ WuP,
                                        bf16* __restrict__ WgP, bf16* __restrict__ WtP) {
  int which = blockIdx.y;                       // 0=Wi 1=Wu 2=Wg 3=Wt
  int ktc   = (which == 2) ? KT_H : KT_D;
  int total = NT_H * ktc * 512;
  for (int idx = blockIdx.x * blockDim.x + threadIdx.x; idx < total;
       idx += gridDim.x * blockDim.x) {
    int e    = idx & 15;
    int lane = (idx >> 4) & 31;
    int tile = idx >> 9;                        // nt * ktc + kt
    int kt   = tile % ktc;
    int nt   = tile / ktc;
    int hf   = lane >> 4;
    int n    = nt * 16 + (lane & 15);
    int k    = kt * 32 + klocal_of(e, hf);
    float val = 0.0f;
    if (n < Hh) {
      if (which == 0)      { if (k < D)  val = Wi[(size_t)k * Hh + n]; }
      else if (which == 1) { if (k < D)  val = Wu[(size_t)k * Hh + n]; }
      else if (which == 2) { if (k < Hh) val = Wg[(size_t)k * Hh + n]; }
      else {
        // concat remap: k in [0,160) -> Ah rows 0..149 ; k in [160,320) -> Hm rows
        int  kk    = (k < HP) ? k : (Hh + (k - HP));
        bool valid = (k < HP) ? (k < Hh) : ((k - HP) < Hh);
        if (valid) val = Wt[(size_t)kk * Hh + n];
      }
    }
    bf16* dst = (which == 0) ? WiP : (which == 1) ? WuP : (which == 2) ? WgP : WtP;
    dst[idx] = (bf16)val;
  }
}

// ---------------------------------------------------------------------------
// Kernel 3: gated projection  Y = sigmoid(X Wi + bi) * tanh(X Wu + bu)
// ---------------------------------------------------------------------------
__global__ void cmp_gate_kernel(const bf16* __restrict__ X,
                                const bf16* __restrict__ WiP,
                                const bf16* __restrict__ WuP,
                                const float* __restrict__ bi,
                                const float* __restrict__ bu,
                                bf16* __restrict__ Y, int ldy) {
  int wave = threadIdx.x >> 5;
  int lane = threadIdx.x & 31;
  int hf   = lane >> 4;
  int row0 = blockIdx.x * 128 + wave * 16;
  for (int nt = 0; nt < NT_H; ++nt) {
    v8f ci = {}; v8f cu = {};
    for (int kt = 0; kt < KT_D; ++kt) {
      v16bf a  = load_frag_rowmajor(X + (size_t)row0 * DP + kt * 32, DP);
      v16bf wi = load_frag_packed(WiP + ((nt * KT_D + kt) << 9));
      v16bf wu = load_frag_packed(WuP + ((nt * KT_D + kt) << 9));
      ci = wmma_bf16(a, wi, ci);
      cu = wmma_bf16(a, wu, cu);
    }
    int col = nt * 16 + (lane & 15);
    float bbi = (col < Hh) ? bi[col] : 0.0f;
    float bbu = (col < Hh) ? bu[col] : 0.0f;
#pragma unroll
    for (int v = 0; v < 8; ++v) {
      int   r  = row0 + v + hf * 8;
      float si = 1.0f / (1.0f + __expf(-(ci[v] + bbi)));
      float tu = tanhf(cu[v] + bbu);
      float g  = (col < Hh) ? si * tu : 0.0f;
      Y[(size_t)r * ldy + col] = (bf16)g;
    }
  }
}

// ---------------------------------------------------------------------------
// Kernel 4: P = Qh Wg + bg   (K = 160)
// ---------------------------------------------------------------------------
__global__ void cmp_linear_kernel(const bf16* __restrict__ X,
                                  const bf16* __restrict__ WP,
                                  const float* __restrict__ bias,
                                  bf16* __restrict__ Y) {
  int wave = threadIdx.x >> 5;
  int lane = threadIdx.x & 31;
  int hf   = lane >> 4;
  int row0 = blockIdx.x * 128 + wave * 16;
  for (int nt = 0; nt < NT_H; ++nt) {
    v8f c = {};
    for (int kt = 0; kt < KT_H; ++kt) {
      v16bf a = load_frag_rowmajor(X + (size_t)row0 * HP + kt * 32, HP);
      v16bf w = load_frag_packed(WP + ((nt * KT_H + kt) << 9));
      c = wmma_bf16(a, w, c);
    }
    int col = nt * 16 + (lane & 15);
    float bb = (col < Hh) ? bias[col] : 0.0f;
#pragma unroll
    for (int v = 0; v < 8; ++v) {
      float o = c[v] + bb;
      Y[(size_t)(row0 + v + hf * 8) * HP + col] = (bf16)((col < Hh) ? o : 0.0f);
    }
  }
}

// ---------------------------------------------------------------------------
// Kernel 5: per-batch transpose Qh [L,HP] -> QhT [HP,L] (LDS tiled)
// ---------------------------------------------------------------------------
__global__ void cmp_transpose_kernel(const bf16* __restrict__ Qh,
                                     bf16* __restrict__ QhT) {
  __shared__ __align__(16) bf16 tile[32][33];
  int b  = blockIdx.z;
  int r0 = blockIdx.x * 32;   // q
  int c0 = blockIdx.y * 32;   // h
  const bf16* src = Qh  + (size_t)b * L * HP;
  bf16*       dst = QhT + (size_t)b * HP * L;
  int tx = threadIdx.x, ty = threadIdx.y;        // blockDim (32, 8)
  for (int i = ty; i < 32; i += 8)
    tile[i][tx] = src[(size_t)(r0 + i) * HP + (c0 + tx)];
  __syncthreads();
  for (int i = ty; i < 32; i += 8)
    dst[(size_t)(c0 + i) * L + (r0 + tx)] = tile[tx][i];
}

// ---------------------------------------------------------------------------
// Kernel 6: fused "flash" attention.
//   S[a,q] = <Ah[a], P[q]> ; softmax over q ; Hm[a] = sum_q prob * Qh[q]
// The 32x160 P key-tile is staged into LDS once per block per q-step by the
// Tensor Data Mover (wave 0 issues a 2D D#), DOUBLE-BUFFERED: tile i+1 is in
// flight (TENSORcnt==1) while tile i is consumed, removing the 8x redundant
// per-wave global reads of P. All waves read B-fragments from LDS.
// Online softmax runs in registers with __shfl_xor in 16-lane halves (wave32).
// ---------------------------------------------------------------------------
__global__ void cmp_flash_kernel(const bf16* __restrict__ Pmat,
                                 bf16* __restrict__ Cat,
                                 const bf16* __restrict__ QhT) {
  __shared__ __align__(16) bf16 p_stage[2][32 * HP];   // 20 KB double buffer
  __shared__ __align__(16) bf16 lds_prob[8][16 * 32];  // per-wave prob scratch
  int wave = threadIdx.x >> 5;
  int lane = threadIdx.x & 31;
  int hf   = lane >> 4;
  int b    = blockIdx.y;
  int a0   = blockIdx.x * 128 + wave * 16;
  int rowA = b * L + a0;

  // Preload Ah A-fragments over the full K (h) range: 5 x 8 VGPRs.
  v16bf afr[KT_H];
#pragma unroll
  for (int kt = 0; kt < KT_H; ++kt)
    afr[kt] = load_frag_rowmajor(Cat + (size_t)rowA * (2 * HP) + kt * 32, 2 * HP);

  v8f acc[NT_H];
#pragma unroll
  for (int nt = 0; nt < NT_H; ++nt) acc[nt] = (v8f){};
  float m[8], l[8];
#pragma unroll
  for (int v = 0; v < 8; ++v) { m[v] = -3.0e38f; l[v] = 0.0f; }

  const bf16* Pb = Pmat + (size_t)b * L * HP;
  const bf16* Tb = QhT  + (size_t)b * HP * L;
  bf16* scr = lds_prob[wave];

#if HAVE_TDM
  if (wave == 0) tdm_load_p_tile(Pb, p_stage[0]);      // prime the pipeline
#endif

  int buf = 0;
  for (int q0 = 0; q0 < L; q0 += 32, buf ^= 1) {
#if HAVE_TDM
    __syncthreads();                      // everyone done reading p_stage[buf^1]
    if (wave == 0) {
      if (q0 + 32 < L) {
        tdm_load_p_tile(Pb + (size_t)(q0 + 32) * HP, p_stage[buf ^ 1]);
        tdm_wait_le1();                   // tile 'buf' complete, next in flight
      } else {
        tdm_wait_le0();
      }
    }
    __syncthreads();                      // p_stage[buf] visible to all waves
    const bf16* Pt = p_stage[buf];
#else
    __syncthreads();
    {
      const unsigned* srcw = (const unsigned*)(Pb + (size_t)q0 * HP);
      unsigned* dstw = (unsigned*)p_stage[0];
      for (int t = threadIdx.x; t < (32 * HP) / 2; t += 256) dstw[t] = srcw[t];
    }
    __syncthreads();
    const bf16* Pt = p_stage[0];
#endif
    // ---- warm next q-block of QhT (value-side) ----
    if (q0 + 32 < L) {
#pragma unroll
      for (int i = 0; i < 5; ++i)
        __builtin_prefetch(Tb + (size_t)(lane * 5 + i) * L + (q0 + 32), 0, 1);
    }
    // ---- S tiles: stage all 10 B-fragments from LDS, then burst 10 WMMAs ----
    v8f s0 = {}; v8f s1 = {};
    {
      v16bf pf0[KT_H], pf1[KT_H];
#pragma unroll
      for (int kt = 0; kt < KT_H; ++kt) {
        pf0[kt] = load_frag_rowmajor(Pt +            kt * 32, HP);
        pf1[kt] = load_frag_rowmajor(Pt + 16 * HP + kt * 32, HP);
      }
#pragma unroll
      for (int kt = 0; kt < KT_H; ++kt) {
        s0 = wmma_bf16(afr[kt], pf0[kt], s0);
        s1 = wmma_bf16(afr[kt], pf1[kt], s1);
      }
    }
    // ---- online softmax: row r = v + hf*8 is spread over 16 lanes ----
    float scale[8];
#pragma unroll
    for (int v = 0; v < 8; ++v) {
      float x = fmaxf(s0[v], s1[v]);
      x = fmaxf(x, __shfl_xor(x, 1, 16));
      x = fmaxf(x, __shfl_xor(x, 2, 16));
      x = fmaxf(x, __shfl_xor(x, 4, 16));
      x = fmaxf(x, __shfl_xor(x, 8, 16));
      float mnew = fmaxf(m[v], x);
      scale[v] = __expf(m[v] - mnew);
      m[v] = mnew;
    }
#pragma unroll
    for (int v = 0; v < 8; ++v) {
      float p0e = __expf(s0[v] - m[v]);
      float p1e = __expf(s1[v] - m[v]);
      float rs = p0e + p1e;
      rs += __shfl_xor(rs, 1, 16);
      rs += __shfl_xor(rs, 2, 16);
      rs += __shfl_xor(rs, 4, 16);
      rs += __shfl_xor(rs, 8, 16);
      l[v] = l[v] * scale[v] + rs;
      int r = v + hf * 8;
      scr[r * 32 +      (lane & 15)] = (bf16)p0e;
      scr[r * 32 + 16 + (lane & 15)] = (bf16)p1e;
#pragma unroll
      for (int nt = 0; nt < NT_H; ++nt) acc[nt][v] *= scale[v];
    }
    // ---- Hm accumulation: prob(16a x 32q) x Qh^T(32q x 16h), 10 n-tiles ----
    v16bf pf = load_frag_rowmajor(scr, 32);
#pragma unroll
    for (int g = 0; g < 2; ++g) {
      v16bf qf[5];
#pragma unroll
      for (int i = 0; i < 5; ++i)
        qf[i] = load_frag_rowmajor(Tb + (size_t)((g * 5 + i) * 16) * L + q0, L);
#pragma unroll
      for (int i = 0; i < 5; ++i)
        acc[g * 5 + i] = wmma_bf16(pf, qf[i], acc[g * 5 + i]);
    }
  }
  // ---- normalize, write Hm into concat buffer high half ----
#pragma unroll
  for (int nt = 0; nt < NT_H; ++nt) {
    int col = nt * 16 + (lane & 15);
#pragma unroll
    for (int v = 0; v < 8; ++v) {
      int   r = rowA + v + hf * 8;
      float o = acc[nt][v] / l[v];
      Cat[(size_t)r * (2 * HP) + HP + col] = (bf16)((col < Hh) ? o : 0.0f);
    }
  }
}

// ---------------------------------------------------------------------------
// Kernel 7: T = relu(concat([Ah, Hm]) Wt + bt)  -> fp32 output [M, 150]
// ---------------------------------------------------------------------------
__global__ void cmp_final_kernel(const bf16* __restrict__ Cat,
                                 const bf16* __restrict__ WtP,
                                 const float* __restrict__ bt,
                                 float* __restrict__ out) {
  int wave = threadIdx.x >> 5;
  int lane = threadIdx.x & 31;
  int hf   = lane >> 4;
  int row0 = blockIdx.x * 128 + wave * 16;
  for (int nt = 0; nt < NT_H; ++nt) {
    v8f c = {};
    for (int kt = 0; kt < KT_D; ++kt) {
      v16bf a = load_frag_rowmajor(Cat + (size_t)row0 * (2 * HP) + kt * 32, 2 * HP);
      v16bf w = load_frag_packed(WtP + ((nt * KT_D + kt) << 9));
      c = wmma_bf16(a, w, c);
    }
    int col = nt * 16 + (lane & 15);
    if (col < Hh) {
      float bb = bt[col];
#pragma unroll
      for (int v = 0; v < 8; ++v) {
        float o = fmaxf(c[v] + bb, 0.0f);
        out[(size_t)(row0 + v + hf * 8) * Hh + col] = o;
      }
    }
  }
}

// ---------------------------------------------------------------------------
// Host launcher
// ---------------------------------------------------------------------------
extern "C" void kernel_launch(void* const* d_in, const int* in_sizes, int n_in,
                              void* d_out, int out_size, void* d_ws, size_t ws_size,
                              hipStream_t stream) {
  const float* Q  = (const float*)d_in[0];
  const float* A  = (const float*)d_in[1];
  const float* Wi = (const float*)d_in[2];
  const float* Wu = (const float*)d_in[3];
  const float* Wg = (const float*)d_in[4];
  const float* Wt = (const float*)d_in[5];
  const float* bi = (const float*)d_in[6];
  const float* bu = (const float*)d_in[7];
  const float* bg = (const float*)d_in[8];
  const float* bt = (const float*)d_in[9];
  float* out = (float*)d_out;

  char* ws = (char*)d_ws;
  size_t off = 0;
  auto alloc = [&](size_t bytes) -> void* {
    void* p = ws + off;
    off = (off + bytes + 255) & ~(size_t)255;
    return p;
  };

  bf16* Qb  = (bf16*)alloc((size_t)M * DP * sizeof(bf16));          // padded bf16 Q
  bf16* Ab  = (bf16*)alloc((size_t)M * DP * sizeof(bf16));          // padded bf16 A
  bf16* Qh  = (bf16*)alloc((size_t)M * HP * sizeof(bf16));          // gate(Q)
  bf16* Pm  = (bf16*)alloc((size_t)M * HP * sizeof(bf16));          // Qh Wg + bg
  bf16* Cat = (bf16*)alloc((size_t)M * 2 * HP * sizeof(bf16));      // [Ah | Hm]
  bf16* QhT = (bf16*)alloc((size_t)M * HP * sizeof(bf16));          // per-batch Qh^T
  bf16* WiP = (bf16*)alloc((size_t)NT_H * KT_D * 512 * sizeof(bf16));
  bf16* WuP = (bf16*)alloc((size_t)NT_H * KT_D * 512 * sizeof(bf16));
  bf16* WgP = (bf16*)alloc((size_t)NT_H * KT_H * 512 * sizeof(bf16));
  bf16* WtP = (bf16*)alloc((size_t)NT_H * KT_D * 512 * sizeof(bf16));

  int convN = M * DP;
  cmp_convert_pad_kernel<<<(convN + 255) / 256, 256, 0, stream>>>(Q, Qb, M);
  cmp_convert_pad_kernel<<<(convN + 255) / 256, 256, 0, stream>>>(A, Ab, M);
  cmp_pack_weights_kernel<<<dim3(64, 4), 256, 0, stream>>>(
      Wi, Wu, Wg, Wt, WiP, WuP, WgP, WtP);

  cmp_gate_kernel<<<M / 128, 256, 0, stream>>>(Qb, WiP, WuP, bi, bu, Qh, HP);
  cmp_gate_kernel<<<M / 128, 256, 0, stream>>>(Ab, WiP, WuP, bi, bu, Cat, 2 * HP);
  cmp_linear_kernel<<<M / 128, 256, 0, stream>>>(Qh, WgP, bg, Pm);
  cmp_transpose_kernel<<<dim3(L / 32, HP / 32, Bb), dim3(32, 8), 0, stream>>>(Qh, QhT);
  cmp_flash_kernel<<<dim3(L / 128, Bb), 256, 0, stream>>>(Pm, Cat, QhT);
  cmp_final_kernel<<<M / 128, 256, 0, stream>>>(Cat, WtP, bt, out);
}
